// WEncoder_34110630265630
// MI455X (gfx1250) — matrix operand order repaired and checked by reading, not compiled, not run on hardware
//
#include <hip/hip_runtime.h>
#include <hip/hip_bf16.h>
#include <math.h>

// ---------------------------------------------------------------------------
// MI455X (gfx1250) implementation of the WEncoder reference.
// GEMM-shaped work (1x1/3x3 convs via implicit GEMM, both FC layers) runs on
// the matrix pipe via V_WMMA_F32_16X16X32_F16 (f16 in, fp32 accumulate).
// f16 activations are kept NHWC so WMMA B fragments are contiguous b128
// loads; f32 activations stay NCHW for GroupNorm / residual / crop.
// Each wave computes a 16(M) x 64(N) strip: the A fragment is loaded once per
// K-chunk and reused across 4 WMMAs.
// ---------------------------------------------------------------------------

typedef _Float16 h16;
typedef __attribute__((ext_vector_type(2)))  _Float16 v2h;
typedef __attribute__((ext_vector_type(8)))  _Float16 v8h;
typedef __attribute__((ext_vector_type(16))) _Float16 v16h;
typedef __attribute__((ext_vector_type(8)))  float    v8f;

__device__ __forceinline__ float lrelu(float v) { return v >= 0.f ? v : 0.2f * v; }

// ---------------------------------------------------------------------------
// Pack fp32 weights [co][ci][tap] -> f16 [co][tap][ci] (tap-major so each
// K-chunk of 32 stays inside one tap; Cin is always a multiple of 32).
// Also used for FC weights (KK==1) with the EqualLinear scale folded in.
// ---------------------------------------------------------------------------
__global__ void pack_w_kernel(const float* __restrict__ S, h16* __restrict__ D,
                              int Cout, int Cin, int KK, float scale) {
  int t = blockIdx.x * blockDim.x + threadIdx.x;
  int total = Cout * Cin * KK;
  if (t >= total) return;
  int co  = t / (Cin * KK);
  int rem = t - co * (Cin * KK);
  int ci  = rem / KK;
  int tap = rem - ci * KK;
  D[(long)co * KK * Cin + (long)tap * Cin + ci] = (h16)(S[t] * scale);
}

// ---------------------------------------------------------------------------
// conv1: 3->32 channels, 3x3, pad 1. Cin=3 is too skinny for WMMA; direct.
// Fused LeakyReLU; writes f32 NCHW and f16 NHWC copies.
// ---------------------------------------------------------------------------
__global__ void conv1_kernel(const float* __restrict__ X, const float* __restrict__ W,
                             float* __restrict__ OutF, h16* __restrict__ OutH) {
  long t = (long)blockIdx.x * blockDim.x + threadIdx.x;
  const long total = 16L * 32 * 32 * 512;
  if (t >= total) return;
  int x  = (int)(t & 511);
  int y  = (int)((t >> 9) & 31);
  int co = (int)((t >> 14) & 31);
  int n  = (int)(t >> 19);
  float acc = 0.f;
  for (int ci = 0; ci < 3; ++ci)
    for (int ky = 0; ky < 3; ++ky) {
      int yy = y + ky - 1;
      if (yy < 0 || yy >= 32) continue;
      for (int kx = 0; kx < 3; ++kx) {
        int xx = x + kx - 1;
        if (xx < 0 || xx >= 512) continue;
        acc += W[((co * 3 + ci) * 3 + ky) * 3 + kx] *
               X[(((long)n * 3 + ci) * 32 + yy) * 512 + xx];
      }
    }
  acc = lrelu(acc);
  OutF[t] = acc;                                          // NCHW
  OutH[(((long)n * 32 + y) * 512 + x) * 32 + co] = (h16)acc;  // NHWC
}

// ---------------------------------------------------------------------------
// Implicit-GEMM conv via V_WMMA_F32_16X16X32_F16.
//   Out[n][co][y][x] = sum_{tap,ci} Wt[co][tap][ci] * X[n][yin][xin][ci]
// X is f16 NHWC; Out is f32 NCHW. One wave = 16(co) x 64(x) strip of one
// output row, 4 accumulators, A fragment reused across the 4 N tiles.
// blockDim=(32,4): 4 waves cover 4 consecutive output rows.
// Fragment layouts used here (A and B share one k order):
//   A (16x32 f16): M=lane&15; elem e: K = e + (hi?16:0)
//   B (32x16 f16): N=lane&15; elem e: K = e + (hi?16:0)   (contiguous in NHWC)
//   C (16x16 f32): N=lane&15; elem v: M = v + (hi?8:0)
// ---------------------------------------------------------------------------
__global__ __launch_bounds__(128)
void conv_wmma_kernel(const h16* __restrict__ X, const h16* __restrict__ Wt,
                      float* __restrict__ Out,
                      int Cin, int Hin, int Win,
                      int Cout, int Hout, int Wout,
                      int stride, int ks) {
  const int lane   = threadIdx.x;
  const int laneHi = lane >> 4;
  const int ln     = lane & 15;
  const int y      = blockIdx.y * 4 + threadIdx.y;
  const int x0     = blockIdx.x * 64;
  const int mtiles = Cout >> 4;
  const int n      = blockIdx.z / mtiles;
  const int co0    = (blockIdx.z % mtiles) << 4;
  const int pad    = (ks == 3) ? 1 : 0;
  const int KK     = ks * ks;

  v8f acc0 = {}, acc1 = {}, acc2 = {}, acc3 = {};
  const long wrow = (long)(co0 + ln) * (KK * Cin) + (laneHi ? 16 : 0);
  const long rowN = ((long)n * Hin) * Win;   // NHWC row base (in pixels)

  for (int tap = 0; tap < KK; ++tap) {
    const int ky  = tap / ks, kx = tap % ks;
    const int yin = y * stride + ky - pad;
    const bool rowok = (yin >= 0) && (yin < Hin);
    const long wbase = wrow + (long)tap * Cin;
    if (tap + 1 < KK) __builtin_prefetch(Wt + wbase + Cin, 0, 1);

    // per-N-tile input bases (NHWC): element = X[base_t + ci]
    long base_t[4];
    bool ok_t[4];
#pragma unroll
    for (int t = 0; t < 4; ++t) {
      const int xin = (x0 + 16 * t + ln) * stride + kx - pad;
      ok_t[t]   = rowok && (xin >= 0) && (xin < Win);
      base_t[t] = (rowN + (long)yin * Win + xin) * Cin + (laneHi ? 16 : 0);
    }

    for (int k0 = 0; k0 < Cin; k0 += 32) {
      v16h a;
#pragma unroll
      for (int v = 0; v < 8; ++v) {
        const int kp = 2 * v;
        v2h p = *(const v2h*)(Wt + wbase + k0 + kp);
        a[2 * v]     = p[0];
        a[2 * v + 1] = p[1];
      }
#pragma unroll
      for (int t = 0; t < 4; ++t) {
        v16h b;
        if (ok_t[t]) {
          v8h lo = *(const v8h*)(X + base_t[t] + k0);
          v8h hi = *(const v8h*)(X + base_t[t] + k0 + 8);
#pragma unroll
          for (int e = 0; e < 8; ++e) { b[e] = lo[e]; b[e + 8] = hi[e]; }
        } else {
#pragma unroll
          for (int e = 0; e < 16; ++e) b[e] = (h16)0;
        }
        v8f* pa = (t == 0) ? &acc0 : (t == 1) ? &acc1 : (t == 2) ? &acc2 : &acc3;
        *pa = __builtin_amdgcn_wmma_f32_16x16x32_f16(false, a, false, b,
                                                     (short)0, *pa, false, false);
      }
    }
  }

  const long oHW   = (long)Hout * Wout;
  const long obase = ((long)n * Cout + co0) * oHW + (long)y * Wout + x0 + ln;
#pragma unroll
  for (int t = 0; t < 4; ++t) {
    const v8f* pa = (t == 0) ? &acc0 : (t == 1) ? &acc1 : (t == 2) ? &acc2 : &acc3;
#pragma unroll
    for (int v = 0; v < 8; ++v) {
      const int m = v + (laneHi ? 8 : 0);
      Out[obase + 16 * t + (long)m * oHW] = (*pa)[v];
    }
  }
}

// ---------------------------------------------------------------------------
// GroupNorm(32 groups, eps=1e-6, no affine) fused with optional residual add,
// LeakyReLU, f32 NCHW store and f16 NHWC store. One workgroup per (n, group).
// HW is always a power of two (lgHW = log2(HW)).
// ---------------------------------------------------------------------------
__global__ __launch_bounds__(256)
void gn_kernel(const float* __restrict__ U, const float* __restrict__ Res,
               float* __restrict__ OutF, h16* __restrict__ OutH,
               int C, int HW, int lgHW) {
  __shared__ float s1[256], s2[256];
  const int  cg   = C >> 5;
  const int  cnt  = cg * HW;
  const int  n    = blockIdx.x >> 5;
  const int  g    = blockIdx.x & 31;
  const long base = ((long)n * C + (long)g * cg) * HW;

  float su = 0.f, sq = 0.f;
  for (int i = threadIdx.x; i < cnt; i += 256) {
    float v = U[base + i];
    su += v;
    sq += v * v;
  }
  s1[threadIdx.x] = su;
  s2[threadIdx.x] = sq;
  __syncthreads();
  for (int st = 128; st > 0; st >>= 1) {
    if (threadIdx.x < st) {
      s1[threadIdx.x] += s1[threadIdx.x + st];
      s2[threadIdx.x] += s2[threadIdx.x + st];
    }
    __syncthreads();
  }
  const float mu  = s1[0] / (float)cnt;
  const float var = s2[0] / (float)cnt - mu * mu;
  const float rs  = rsqrtf(var + 1e-6f);

  const long nBase16 = (long)n * HW * C;
  for (int i = threadIdx.x; i < cnt; i += 256) {
    float v = (U[base + i] - mu) * rs;
    if (Res) v += Res[base + i];
    v = lrelu(v);
    if (OutF) OutF[base + i] = v;
    if (OutH) {
      const int ch = g * cg + (i >> lgHW);
      const int p  = i & (HW - 1);
      OutH[nBase16 + (long)p * C + ch] = (h16)v;   // NHWC
    }
  }
}

// ---------------------------------------------------------------------------
// Crop 4-column windows around locs//32 from feat [16,512,4,64] (f32 NCHW),
// flatten in (c,h,w) order to 8192, PixelNorm, store f16 rows [384][8192].
// One workgroup per (b, nc) row; gather recomputed in pass 2 (L2-hot).
// ---------------------------------------------------------------------------
__global__ __launch_bounds__(256)
void crop_kernel(const float* __restrict__ F, const int* __restrict__ locs,
                 h16* __restrict__ V) {
  __shared__ float sred[256];
  const int r      = blockIdx.x;           // 0..383  (b*24 + nc)
  const int b      = r / 24;
  const int center = locs[r] / 32;         // locs // (4*DOWN_H), locs >= 0

  float sq = 0.f;
  for (int i = threadIdx.x; i < 8192; i += 256) {
    int wi = i & 3, h = (i >> 2) & 3, c = i >> 4;
    int col = center - 2 + wi;
    float v = 0.f;
    if (col >= 0 && col < 64) v = F[(((long)b * 512 + c) * 4 + h) * 64 + col];
    sq += v * v;
  }
  sred[threadIdx.x] = sq;
  __syncthreads();
  for (int st = 128; st > 0; st >>= 1) {
    if (threadIdx.x < st) sred[threadIdx.x] += sred[threadIdx.x + st];
    __syncthreads();
  }
  const float rs = rsqrtf(sred[0] / 8192.f + 1e-8f);

  for (int i = threadIdx.x; i < 8192; i += 256) {
    int wi = i & 3, h = (i >> 2) & 3, c = i >> 4;
    int col = center - 2 + wi;
    float v = 0.f;
    if (col >= 0 && col < 64) v = F[(((long)b * 512 + c) * 4 + h) * 64 + col];
    V[(long)r * 8192 + i] = (h16)(v * rs);
  }
}

// ---------------------------------------------------------------------------
// Dense layer via WMMA: out[r][m] = lrelu(sum_k Inp[r][k]*Wt[m][k] + bias)*sqrt2
// (optionally masked by locs<2048). One wave = 16(m) x 64(rows), A fragment
// reused across 4 row tiles; B fragments are contiguous b128 loads.
// ---------------------------------------------------------------------------
__global__ __launch_bounds__(32)
void fc_wmma_kernel(const h16* __restrict__ Wt,   // [M][K] f16, scale folded
                    const h16* __restrict__ Inp,  // [rows][K] f16
                    const float* __restrict__ bias,
                    h16* __restrict__ OutH,       // [rows][M] or null
                    float* __restrict__ OutF,     // [rows][M] or null
                    const int* __restrict__ locs, // null or [rows]
                    int M, int K) {
  const int lane   = threadIdx.x;
  const int laneHi = lane >> 4;
  const int ln     = lane & 15;
  const int r0     = blockIdx.x << 6;     // 64 rows per wave
  const int m0     = blockIdx.y << 4;

  v8f acc0 = {}, acc1 = {}, acc2 = {}, acc3 = {};
  const h16* wrow = Wt + (long)(m0 + ln) * K + (laneHi ? 16 : 0);
  const h16* brow0 = Inp + (long)(r0 + ln) * K + (laneHi ? 16 : 0);

  for (int k0 = 0; k0 < K; k0 += 32) {
    v16h a;
#pragma unroll
    for (int v = 0; v < 8; ++v) {
      const int kp = 2 * v;
      v2h p = *(const v2h*)(wrow + k0 + kp);
      a[2 * v]     = p[0];
      a[2 * v + 1] = p[1];
    }
#pragma unroll
    for (int t = 0; t < 4; ++t) {
      const h16* brow = brow0 + (long)(16 * t) * K;
      v8h lo = *(const v8h*)(brow + k0);
      v8h hi = *(const v8h*)(brow + k0 + 8);
      v16h b;
#pragma unroll
      for (int e = 0; e < 8; ++e) { b[e] = lo[e]; b[e + 8] = hi[e]; }
      v8f* pa = (t == 0) ? &acc0 : (t == 1) ? &acc1 : (t == 2) ? &acc2 : &acc3;
      *pa = __builtin_amdgcn_wmma_f32_16x16x32_f16(false, a, false, b,
                                                   (short)0, *pa, false, false);
    }
  }

  const float sq2 = 1.41421356237309515f;
#pragma unroll
  for (int t = 0; t < 4; ++t) {
    const int r = r0 + 16 * t + ln;
    float vm = 1.f;
    if (locs) vm = (locs[r] < 2048) ? 1.f : 0.f;
    const v8f* pa = (t == 0) ? &acc0 : (t == 1) ? &acc1 : (t == 2) ? &acc2 : &acc3;
#pragma unroll
    for (int v = 0; v < 8; ++v) {
      const int m = m0 + v + (laneHi ? 8 : 0);
      float val = (*pa)[v] + bias[m];
      val = lrelu(val) * sq2 * vm;
      if (OutH) OutH[(long)r * M + m] = (h16)val;
      if (OutF) OutF[(long)r * M + m] = val;
    }
  }
}

// ---------------------------------------------------------------------------
// Host orchestration.
// Assumed d_in order (jax pytree flatten, sorted dict keys):
//   0:x 1:locs 2:conv1 3:fc1_b 4:fc1_w 5:fc2_b 6:fc2_w
//   then layers stage0..4, each block: w1, w2, (wd iff first block of stage).
// ---------------------------------------------------------------------------
extern "C" void kernel_launch(void* const* d_in, const int* in_sizes, int n_in,
                              void* d_out, int out_size, void* d_ws, size_t ws_size,
                              hipStream_t stream) {
  (void)in_sizes; (void)n_in; (void)out_size; (void)ws_size;

  const float* x      = (const float*)d_in[0];
  const int*   locs   = (const int*)d_in[1];
  const float* conv1w = (const float*)d_in[2];
  const float* fc1b   = (const float*)d_in[3];
  const float* fc1w   = (const float*)d_in[4];
  const float* fc2b   = (const float*)d_in[5];
  const float* fc2w   = (const float*)d_in[6];

  static const int PLANES [5] = {32, 64, 128, 256, 512};
  static const int NBLK   [5] = {3, 4, 6, 6, 3};
  static const int STRIDE [5] = {2, 1, 2, 1, 2};

  // ---- workspace carve (static worst-case offsets) ----
  char* ws = (char*)d_ws;
  h16*  WH = (h16*)ws;                         // f16 packed weights (<=64MB)
  size_t off = 64ull << 20;
  float* A0 = (float*)(ws + off); off += 36ull << 20;
  float* A1 = (float*)(ws + off); off += 36ull << 20;
  float* U  = (float*)(ws + off); off += 36ull << 20;
  float* R  = (float*)(ws + off); off += 20ull << 20;
  h16* A0h = (h16*)(ws + off); off += 18ull << 20;
  h16* A1h = (h16*)(ws + off); off += 18ull << 20;
  h16* Th  = (h16*)(ws + off); off += 18ull << 20;
  h16* Vh  = (h16*)(ws + off); off += 8ull  << 20;
  h16* Hh  = (h16*)(ws + off); off += 1ull  << 20;

  // ---- pack FC weights (EqualLinear scales folded in) ----
  size_t whoff = 0;
  const float s1 = 1.0f / sqrtf(8192.0f);
  const float s2 = 1.0f / sqrtf(512.0f);
  h16* fc1wh = WH + whoff; whoff += (size_t)512 * 8192;
  pack_w_kernel<<<(512 * 8192 + 255) / 256, 256, 0, stream>>>(fc1w, fc1wh, 512, 8192, 1, s1);
  h16* fc2wh = WH + whoff; whoff += (size_t)512 * 512;
  pack_w_kernel<<<(512 * 512 + 255) / 256, 256, 0, stream>>>(fc2w, fc2wh, 512, 512, 1, s2);

  // ---- stem ----
  {
    const long total = 16L * 32 * 32 * 512;
    conv1_kernel<<<(int)((total + 255) / 256), 256, 0, stream>>>(x, conv1w, A0, A0h);
  }

  // ---- residual stages ----
  float* curF = A0; h16* curH = A0h;
  float* nxtF = A1; h16* nxtH = A1h;
  int Cin = 32, H = 32, W = 512;
  int din = 7;
  const dim3 blk(32, 4);

  for (int s = 0; s < 5; ++s) {
    const int planes = PLANES[s];
    for (int bi = 0; bi < NBLK[s]; ++bi) {
      const int  stride = (bi == 0) ? STRIDE[s] : 1;
      const bool haswd  = (bi == 0);
      const float* w1 = (const float*)d_in[din++];
      const float* w2 = (const float*)d_in[din++];
      const float* wd = haswd ? (const float*)d_in[din++] : nullptr;

      // pack this block's weights
      h16* w1h = WH + whoff; whoff += (size_t)planes * Cin;
      pack_w_kernel<<<(planes * Cin + 255) / 256, 256, 0, stream>>>(w1, w1h, planes, Cin, 1, 1.f);
      h16* w2h = WH + whoff; whoff += (size_t)planes * planes * 9;
      pack_w_kernel<<<(planes * planes * 9 + 255) / 256, 256, 0, stream>>>(w2, w2h, planes, planes, 9, 1.f);
      h16* wdh = nullptr;
      if (haswd) {
        wdh = WH + whoff; whoff += (size_t)planes * Cin;
        pack_w_kernel<<<(planes * Cin + 255) / 256, 256, 0, stream>>>(wd, wdh, planes, Cin, 1, 1.f);
      }

      const int Ho = H / stride, Wo = W / stride;
      const int lgHW  = __builtin_ctz(H * W);
      const int lgHWo = __builtin_ctz(Ho * Wo);

      // 1x1 conv -> U at (planes, H, W)
      conv_wmma_kernel<<<dim3(W / 64, H / 4, 16 * (planes / 16)), blk, 0, stream>>>(
          curH, w1h, U, Cin, H, W, planes, H, W, 1, 1);
      // GN + lrelu -> Th (f16 NHWC only)
      gn_kernel<<<16 * 32, 256, 0, stream>>>(U, nullptr, nullptr, Th, planes, H * W, lgHW);
      // 3x3 conv (stride) -> U at (planes, Ho, Wo)
      conv_wmma_kernel<<<dim3(Wo / 64, Ho / 4, 16 * (planes / 16)), blk, 0, stream>>>(
          Th, w2h, U, planes, H, W, planes, Ho, Wo, stride, 3);
      // residual path
      const float* res;
      if (haswd) {
        conv_wmma_kernel<<<dim3(Wo / 64, Ho / 4, 16 * (planes / 16)), blk, 0, stream>>>(
            curH, wdh, R, Cin, H, W, planes, Ho, Wo, stride, 1);
        res = R;
      } else {
        res = curF;   // identity (stride==1, same channels)
      }
      // GN + residual + lrelu -> next activation (f32 NCHW + f16 NHWC)
      gn_kernel<<<16 * 32, 256, 0, stream>>>(U, res, nxtF, nxtH, planes, Ho * Wo, lgHWo);

      { float* t = curF; curF = nxtF; nxtF = t; }
      { h16*   t = curH; curH = nxtH; nxtH = t; }
      Cin = planes; H = Ho; W = Wo;
    }
  }

  // ---- crop + PixelNorm -> Vh [384][8192] f16 ----
  crop_kernel<<<384, 256, 0, stream>>>(curF, locs, Vh);

  // ---- fc1: 8192 -> 512, fused bias + lrelu*sqrt2 -> Hh [384][512] f16 ----
  fc_wmma_kernel<<<dim3(6, 32), 32, 0, stream>>>(fc1wh, Vh, fc1b, Hh, nullptr, nullptr, 512, 8192);
  // ---- fc2: 512 -> 512, fused bias + lrelu*sqrt2 + validity mask -> d_out ----
  fc_wmma_kernel<<<dim3(6, 32), 32, 0, stream>>>(fc2wh, Hh, fc2b, nullptr, (float*)d_out, locs, 512, 512);
}
